// OACLIPv3_56221121905262
// MI455X (gfx1250) — matrix operand order repaired and checked by reading, not compiled, not run on hardware
//
#include <hip/hip_runtime.h>
#include <hip/hip_bf16.h>
#include <math.h>

typedef _Float16 f16;
typedef __attribute__((ext_vector_type(16))) _Float16 v16h;
typedef __attribute__((ext_vector_type(8)))  float    v8f;

#define WMMA_F16(A, B, C) \
  __builtin_amdgcn_wmma_f32_16x16x32_f16(false, (A), false, (B), (short)0, (C), false, false)

static constexpr int INP = 2048;
static constexpr int EMB = 512;
static constexpr int H   = 2;
static constexpr int BSZ = 16;
static constexpr int LL  = 1024;
static constexpr float SQRT_LAMBDA = 3.16227766016838f; // sqrt(10)

// ---------------------------------------------------------------------------
// Fragment loaders (ISA 7.12.2 wave32 layouts):
// A 16x32 f16: lane m=lane&15, half=lane>>4. Elements 0..7 <- K = half*8 + 0..7,
//              elements 8..15 <- K = 16 + half*8 + 0..7 (two contiguous 16B runs).
// B 32x16 f16: lane n=lane&15, half=lane>>4. Same K order as A (half*8 + 0..7, 16 + half*8 + 0..7).
// ---------------------------------------------------------------------------
__device__ __forceinline__ v16h frag_a(const f16* __restrict__ base, int ld) {
  int lane = threadIdx.x & 31;
  int m = lane & 15, half = lane >> 4;
  const f16* p = base + m * ld + half * 8;
  v16h r;
#pragma unroll
  for (int e = 0; e < 8; ++e) { r[e] = p[e]; r[8 + e] = p[16 + e]; }
  return r;
}

__device__ __forceinline__ v16h frag_b(const f16* __restrict__ base, int ld) {
  int lane = threadIdx.x & 31;
  int n = lane & 15, half = lane >> 4;
  const f16* p = base + n * ld + half * 8;
  v16h r;
#pragma unroll
  for (int e = 0; e < 8; ++e) { r[e] = p[e]; r[8 + e] = p[16 + e]; }
  return r;
}

// ---------------------------------------------------------------------------
// Kernel 0: transpose+cast W (H, INP, EMB) f32 -> WhT (H, EMB, INP) f16.
// 4 MB output, done once; makes every GEMM staging load K-contiguous.
// grid = (EMB/32, INP/32, H), block = 256 (32x8).
// ---------------------------------------------------------------------------
__global__ __launch_bounds__(256) void wcast_kernel(
    const float* __restrict__ Wt, f16* __restrict__ WhT)
{
  __shared__ float tile[32][33];
  const int h  = blockIdx.z;
  const int e0 = blockIdx.x * 32;
  const int k0 = blockIdx.y * 32;
  const int tx = threadIdx.x & 31, ty = threadIdx.x >> 5;

  const float* src = Wt + (size_t)h * INP * EMB;
#pragma unroll
  for (int i = 0; i < 4; ++i)
    tile[ty + i * 8][tx] = src[(size_t)(k0 + ty + i * 8) * EMB + e0 + tx];
  __syncthreads();
  f16* dst = WhT + (size_t)h * EMB * INP;
#pragma unroll
  for (int i = 0; i < 4; ++i)
    dst[(size_t)(e0 + ty + i * 8) * INP + k0 + tx] = (f16)tile[tx][ty + i * 8];
}

// ---------------------------------------------------------------------------
// Kernel 1: fused projection GEMM + bias + L2 normalize + *sqrt(10), f16 out.
// Block = 32 rows (L) x 512 cols (full EMB) so row norms are block-local.
// 256 threads = 8 waves; wave w owns 64 columns; 2x4 tiles of 16x16 per wave.
// gridDim.x = H*BS*(L/32) = 1024.
// ---------------------------------------------------------------------------
__global__ __launch_bounds__(256) void proj_kernel(
    const float* __restrict__ img,   // (BS, INP, L) f32
    const f16*   __restrict__ WhT,   // (H, EMB, INP) f16 (pre-transposed)
    const float* __restrict__ bias,  // (H, EMB)
    f16* __restrict__ Qn)            // (H*BS, L, EMB) f16
{
  constexpr int ldA = 40, ldB = 40;
  __shared__ f16 As[32 * ldA];
  __shared__ f16 Bs[EMB * ldB];
  __shared__ float rowsq[32];

  const int rblk = blockIdx.x & 31;
  const int hb   = blockIdx.x >> 5;
  const int h    = hb >> 4;
  const int b    = hb & 15;
  const int l0   = rblk * 32;
  const int tid  = threadIdx.x;
  const int lane = tid & 31;
  const int wave = tid >> 5;
  const int n    = lane & 15;
  const int half = lane >> 4;

  const float* imgB = img + (size_t)b * INP * LL;
  const f16*   WhTh = WhT + (size_t)h * EMB * INP;

  // this thread's fixed staging coordinates
  const int ak = tid >> 5, al = tid & 31;        // A: 4 elems at k = ak + 8*it

  v8f zf = {0.f, 0.f, 0.f, 0.f, 0.f, 0.f, 0.f, 0.f};
  v8f acc[2][4];
#pragma unroll
  for (int i = 0; i < 2; ++i)
#pragma unroll
    for (int j = 0; j < 4; ++j) acc[i][j] = zf;

  for (int k0 = 0; k0 < INP; k0 += 32) {
    // A tile: batch the 4 global loads into regs first (one wait), then store.
    float ar[4];
#pragma unroll
    for (int it = 0; it < 4; ++it)
      ar[it] = imgB[(size_t)(k0 + ak + it * 8) * LL + l0 + al];
#pragma unroll
    for (int it = 0; it < 4; ++it)
      As[al * ldA + ak + it * 8] = (f16)ar[it];

    // B tile: K-contiguous f16 copies from pre-transposed W (b128 in/out).
#pragma unroll
    for (int c = 0; c < 2; ++c) {
      const int e = tid + c * 256;
      const f16* src = WhTh + (size_t)e * INP + k0;
      f16* dst = Bs + e * ldB;
#pragma unroll
      for (int u = 0; u < 32; ++u) dst[u] = src[u];
    }
    if (k0 + 32 < INP) {
      __builtin_prefetch(imgB + (size_t)(k0 + 32 + ak) * LL + l0 + al, 0, 1);
      __builtin_prefetch(WhTh + (size_t)tid * INP + k0 + 32, 0, 1);
    }
    __syncthreads();

    v16h a0 = frag_a(As, ldA);
    v16h a1 = frag_a(As + 16 * ldA, ldA);
#pragma unroll
    for (int j = 0; j < 4; ++j) {
      v16h bj = frag_b(Bs + (wave * 64 + j * 16) * ldB, ldB);
      acc[0][j] = WMMA_F16(a0, bj, acc[0][j]);
      acc[1][j] = WMMA_F16(a1, bj, acc[1][j]);
    }
    __syncthreads();
  }

  // bias add (C layout: element r of lane -> row = 8*half + r, col = n)
#pragma unroll
  for (int j = 0; j < 4; ++j) {
    float bv = bias[h * EMB + wave * 64 + j * 16 + n];
#pragma unroll
    for (int i = 0; i < 2; ++i)
#pragma unroll
      for (int r = 0; r < 8; ++r) acc[i][j][r] += bv;
  }

  if (tid < 32) rowsq[tid] = 0.f;
  __syncthreads();

  // per-row sum of squares: in-register over j, shuffle over the 16 col-lanes,
  // LDS atomic across the 8 waves (each wave owns a 64-col strip of same rows).
#pragma unroll
  for (int i = 0; i < 2; ++i)
#pragma unroll
    for (int r = 0; r < 8; ++r) {
      float s = 0.f;
#pragma unroll
      for (int j = 0; j < 4; ++j) s += acc[i][j][r] * acc[i][j][r];
#pragma unroll
      for (int off = 1; off < 16; off <<= 1) s += __shfl_xor(s, off, 32);
      if (n == 0) atomicAdd(&rowsq[i * 16 + half * 8 + r], s);
    }
  __syncthreads();

  f16* out = Qn + ((size_t)hb * LL + l0) * EMB;
#pragma unroll
  for (int i = 0; i < 2; ++i)
#pragma unroll
    for (int r = 0; r < 8; ++r) {
      int row = i * 16 + half * 8 + r;
      float scale = SQRT_LAMBDA / fmaxf(sqrtf(rowsq[row]), 1e-12f);
#pragma unroll
      for (int j = 0; j < 4; ++j) {
        int col = wave * 64 + j * 16 + n;
        out[(size_t)row * EMB + col] = (f16)(acc[i][j][r] * scale);
      }
    }
}

// ---------------------------------------------------------------------------
// Shared rel-tile GEMM: 64 (l) x 128 (m) tile, K = 512, f16 WMMA, f32 acc.
// 8 waves arranged 2x4; each wave computes a 32x32 sub-tile (2x2 WMMA tiles).
// Ping-pong double-buffered LDS: one barrier per K-step, loads overlap WMMA.
// ---------------------------------------------------------------------------
static constexpr int RLDA = 40, RLDB = 40;

__device__ __forceinline__ void stage_tiles(
    const f16* __restrict__ Q1, const f16* __restrict__ Q2,
    int l0, int m0, int k0, f16* __restrict__ A, f16* __restrict__ B)
{
  const int tid = threadIdx.x;
  {
    int l = tid >> 2, koff = (tid & 3) * 8;
    const f16* src = Q1 + (size_t)(l0 + l) * EMB + k0 + koff;
    f16* dst = A + l * RLDA + koff;
#pragma unroll
    for (int e = 0; e < 8; ++e) dst[e] = src[e];
  }
#pragma unroll 2
  for (int idx = tid; idx < 512; idx += 256) {
    int m = idx >> 2, koff = (idx & 3) * 8;
    const f16* src = Q2 + (size_t)(m0 + m) * EMB + k0 + koff;
    f16* dst = B + m * RLDB + koff;
#pragma unroll
    for (int e = 0; e < 8; ++e) dst[e] = src[e];
  }
  if (k0 + 32 < EMB) {
    __builtin_prefetch(Q1 + (size_t)(l0 + (tid >> 2)) * EMB + k0 + 32, 0, 1);
    __builtin_prefetch(Q2 + (size_t)(m0 + (tid >> 1)) * EMB + k0 + 32, 0, 1);
  }
}

__device__ __forceinline__ void rel_tile(
    const f16* __restrict__ Q1, const f16* __restrict__ Q2,
    int l0, int m0, f16* AsLDS /*[2][64*RLDA]*/, f16* BsLDS /*[2][128*RLDB]*/,
    v8f acc[2][2])
{
  const int wave = threadIdx.x >> 5;
  const int wi   = wave >> 2;  // 0..1 -> row half
  const int wj   = wave & 3;   // 0..3 -> col quarter

  v8f zf = {0.f, 0.f, 0.f, 0.f, 0.f, 0.f, 0.f, 0.f};
  acc[0][0] = zf; acc[0][1] = zf; acc[1][0] = zf; acc[1][1] = zf;

  stage_tiles(Q1, Q2, l0, m0, 0, AsLDS, BsLDS);
  int cur = 0;
  for (int k0 = 0; k0 < EMB; k0 += 32) {
    __syncthreads();  // current buffer's stores (prev iter) are visible
    if (k0 + 32 < EMB)
      stage_tiles(Q1, Q2, l0, m0, k0 + 32,
                  AsLDS + (cur ^ 1) * (64 * RLDA),
                  BsLDS + (cur ^ 1) * (128 * RLDB));
    const f16* Ab = AsLDS + cur * (64 * RLDA);
    const f16* Bb = BsLDS + cur * (128 * RLDB);
    v16h a0 = frag_a(Ab + (wi * 32) * RLDA, RLDA);
    v16h a1 = frag_a(Ab + (wi * 32 + 16) * RLDA, RLDA);
    v16h b0 = frag_b(Bb + (wj * 32) * RLDB, RLDB);
    v16h b1 = frag_b(Bb + (wj * 32 + 16) * RLDB, RLDB);
    acc[0][0] = WMMA_F16(a0, b0, acc[0][0]);
    acc[0][1] = WMMA_F16(a0, b1, acc[0][1]);
    acc[1][0] = WMMA_F16(a1, b0, acc[1][0]);
    acc[1][1] = WMMA_F16(a1, b1, acc[1][1]);
    cur ^= 1;
  }
  __syncthreads();
}

// ---------------------------------------------------------------------------
// Kernel 2: exp-sum stats. rel in [-10,10] so exp is safe without max-sub.
// grid = (L/64, L/128, H*BS)
// ---------------------------------------------------------------------------
__global__ __launch_bounds__(256) void stats_kernel(
    const f16* __restrict__ Qn1, const f16* __restrict__ Qn2,
    float* __restrict__ row_sum, float* __restrict__ col_sum)
{
  __shared__ f16 As[2 * 64 * RLDA];
  __shared__ f16 Bs[2 * 128 * RLDB];
  __shared__ float rs[64], cs[128];

  const int hb = blockIdx.z;
  const int l0 = blockIdx.x * 64;
  const int m0 = blockIdx.y * 128;
  const int tid = threadIdx.x, lane = tid & 31, wave = tid >> 5;
  const int wi = wave >> 2, wj = wave & 3;
  const int n = lane & 15, half = lane >> 4;

  const f16* Q1 = Qn1 + (size_t)hb * LL * EMB;
  const f16* Q2 = Qn2 + (size_t)hb * LL * EMB;

  v8f acc[2][2];
  rel_tile(Q1, Q2, l0, m0, As, Bs, acc);

  float ev[2][2][8];
#pragma unroll
  for (int ti = 0; ti < 2; ++ti)
#pragma unroll
    for (int tj = 0; tj < 2; ++tj)
#pragma unroll
      for (int r = 0; r < 8; ++r) ev[ti][tj][r] = __expf(acc[ti][tj][r]);

  if (tid < 64) rs[tid] = 0.f;
  if (tid < 128) cs[tid] = 0.f;
  __syncthreads();

  // column sums: sum over ti,r in-register, over halves via shfl_xor(16)
#pragma unroll
  for (int tj = 0; tj < 2; ++tj) {
    float s = 0.f;
#pragma unroll
    for (int ti = 0; ti < 2; ++ti)
#pragma unroll
      for (int r = 0; r < 8; ++r) s += ev[ti][tj][r];
    s += __shfl_xor(s, 16, 32);
    if (lane < 16) atomicAdd(&cs[wj * 32 + tj * 16 + n], s);
  }
  // row sums: sum over tj in-register, over 16 col-lanes via butterfly
#pragma unroll
  for (int ti = 0; ti < 2; ++ti)
#pragma unroll
    for (int r = 0; r < 8; ++r) {
      float s = ev[ti][0][r] + ev[ti][1][r];
#pragma unroll
      for (int off = 1; off < 16; off <<= 1) s += __shfl_xor(s, off, 32);
      if (n == 0) atomicAdd(&rs[wi * 32 + ti * 16 + half * 8 + r], s);
    }
  __syncthreads();

  if (tid < 64)  atomicAdd(&row_sum[(size_t)hb * LL + l0 + tid], rs[tid]);
  if (tid < 128) atomicAdd(&col_sum[(size_t)hb * LL + m0 + tid], cs[tid]);
}

// ---------------------------------------------------------------------------
// Kernel 3: recompute rel tile, accumulate softmax-sum weights.
// sim12[m] += sum_l e/rowsum[l] ; sim21[l] += sum_m e/colsum[m]
// ---------------------------------------------------------------------------
__global__ __launch_bounds__(256) void sim_kernel(
    const f16* __restrict__ Qn1, const f16* __restrict__ Qn2,
    const float* __restrict__ row_sum, const float* __restrict__ col_sum,
    float* __restrict__ sim21_raw, float* __restrict__ sim12_raw)
{
  __shared__ f16 As[2 * 64 * RLDA];
  __shared__ f16 Bs[2 * 128 * RLDB];
  __shared__ float rinv[64], cinv[128], racc[64], cacc[128];

  const int hb = blockIdx.z;
  const int l0 = blockIdx.x * 64;
  const int m0 = blockIdx.y * 128;
  const int tid = threadIdx.x, lane = tid & 31, wave = tid >> 5;
  const int wi = wave >> 2, wj = wave & 3;
  const int n = lane & 15, half = lane >> 4;

  const f16* Q1 = Qn1 + (size_t)hb * LL * EMB;
  const f16* Q2 = Qn2 + (size_t)hb * LL * EMB;

  v8f acc[2][2];
  rel_tile(Q1, Q2, l0, m0, As, Bs, acc);

  if (tid < 64)  { rinv[tid] = 1.f / row_sum[(size_t)hb * LL + l0 + tid]; racc[tid] = 0.f; }
  if (tid < 128) { cinv[tid] = 1.f / col_sum[(size_t)hb * LL + m0 + tid]; cacc[tid] = 0.f; }
  __syncthreads();

  float ev[2][2][8];
#pragma unroll
  for (int ti = 0; ti < 2; ++ti)
#pragma unroll
    for (int tj = 0; tj < 2; ++tj)
#pragma unroll
      for (int r = 0; r < 8; ++r) ev[ti][tj][r] = __expf(acc[ti][tj][r]);

  // sim12 contribution: per column, sum over local rows of e * rinv[row]
#pragma unroll
  for (int tj = 0; tj < 2; ++tj) {
    float s = 0.f;
#pragma unroll
    for (int ti = 0; ti < 2; ++ti)
#pragma unroll
      for (int r = 0; r < 8; ++r)
        s += ev[ti][tj][r] * rinv[wi * 32 + ti * 16 + half * 8 + r];
    s += __shfl_xor(s, 16, 32);
    if (lane < 16) atomicAdd(&cacc[wj * 32 + tj * 16 + n], s);
  }
  // sim21 contribution: per row, sum over local cols of e * cinv[col]
#pragma unroll
  for (int ti = 0; ti < 2; ++ti)
#pragma unroll
    for (int r = 0; r < 8; ++r) {
      float s = ev[ti][0][r] * cinv[wj * 32 + n]
              + ev[ti][1][r] * cinv[wj * 32 + 16 + n];
#pragma unroll
      for (int off = 1; off < 16; off <<= 1) s += __shfl_xor(s, off, 32);
      if (n == 0) atomicAdd(&racc[wi * 32 + ti * 16 + half * 8 + r], s);
    }
  __syncthreads();

  if (tid < 64)  atomicAdd(&sim21_raw[(size_t)hb * LL + l0 + tid], racc[tid]);
  if (tid < 128) atomicAdd(&sim12_raw[(size_t)hb * LL + m0 + tid], cacc[tid]);
}

// ---------------------------------------------------------------------------
// Kernel 4: per-(h,b) renormalize and write (2, H, BS, L) output.
// 64 blocks: which = blk>>5 (0=sim21, 1=sim12), hb = blk&31.
// ---------------------------------------------------------------------------
__global__ __launch_bounds__(256) void finalize_kernel(
    const float* __restrict__ sim21_raw, const float* __restrict__ sim12_raw,
    float* __restrict__ out)
{
  __shared__ float red[256];
  const int which = blockIdx.x >> 5;
  const int hb    = blockIdx.x & 31;
  const int tid   = threadIdx.x;
  const float* src = (which == 0 ? sim21_raw : sim12_raw) + (size_t)hb * LL;

  float s = 0.f;
  for (int i = tid; i < LL; i += 256) s += src[i];
  red[tid] = s;
  __syncthreads();
  for (int off = 128; off > 0; off >>= 1) {
    if (tid < off) red[tid] += red[tid + off];
    __syncthreads();
  }
  float inv = 1.f / (red[0] + 1e-8f);
  float* dst = out + (size_t)which * H * BSZ * LL + (size_t)hb * LL;
  for (int i = tid; i < LL; i += 256) dst[i] = src[i] * inv;
}

__global__ void zero_kernel(float* __restrict__ p, int nElem) {
  int i = blockIdx.x * 256 + threadIdx.x;
  if (i < nElem) p[i] = 0.f;
}

// ---------------------------------------------------------------------------
extern "C" void kernel_launch(void* const* d_in, const int* in_sizes, int n_in,
                              void* d_out, int out_size, void* d_ws, size_t ws_size,
                              hipStream_t stream) {
  const float* img1 = (const float*)d_in[0];
  const float* img2 = (const float*)d_in[1];
  const float* Wt   = (const float*)d_in[2];
  const float* bias = (const float*)d_in[3];
  float* out = (float*)d_out;

  char* ws = (char*)d_ws;
  const size_t qbytes = (size_t)H * BSZ * LL * EMB * sizeof(f16); // 32 MiB each
  const size_t wbytes = (size_t)H * EMB * INP * sizeof(f16);      // 4 MiB
  f16* Qn1 = (f16*)ws;
  f16* Qn2 = (f16*)(ws + qbytes);
  f16* WhT = (f16*)(ws + 2 * qbytes);
  float* row_sum   = (float*)(ws + 2 * qbytes + wbytes);
  float* col_sum   = row_sum + (size_t)H * BSZ * LL;
  float* sim21_raw = col_sum + (size_t)H * BSZ * LL;
  float* sim12_raw = sim21_raw + (size_t)H * BSZ * LL;

  // zero the four contiguous stat arrays (4 * 32 * 1024 floats)
  const int nstat = 4 * H * BSZ * LL;
  zero_kernel<<<(nstat + 255) / 256, 256, 0, stream>>>(row_sum, nstat);

  // one-time W transpose+cast to f16 (H, EMB, INP)
  dim3 gw(EMB / 32, INP / 32, H);
  wcast_kernel<<<gw, 256, 0, stream>>>(Wt, WhT);

  // projection + bias + normalize (fold sqrt(LAMBDA) into both operands)
  proj_kernel<<<H * BSZ * (LL / 32), 256, 0, stream>>>(img1, WhT, bias, Qn1);
  proj_kernel<<<H * BSZ * (LL / 32), 256, 0, stream>>>(img2, WhT, bias, Qn2);

  dim3 gs(LL / 64, LL / 128, H * BSZ);
  stats_kernel<<<gs, 256, 0, stream>>>(Qn1, Qn2, row_sum, col_sum);
  sim_kernel<<<gs, 256, 0, stream>>>(Qn1, Qn2, row_sum, col_sum, sim21_raw, sim12_raw);

  finalize_kernel<<<2 * H * BSZ, 256, 0, stream>>>(sim21_raw, sim12_raw, out);
}
